// SpatialAttention_88974542504709
// MI455X (gfx1250) — matrix operand order repaired and checked by reading, not compiled, not run on hardware
//
#include <hip/hip_runtime.h>

#define NB   8
#define NC   64
#define NR   8
#define NPIX 4096   // 64*64
#define LOG2E 1.4426950408889634f

typedef __attribute__((ext_vector_type(16))) _Float16 v16h;
typedef __attribute__((ext_vector_type(8)))  _Float16 v8h;
typedef __attribute__((ext_vector_type(8)))  float    v8f;
typedef __attribute__((ext_vector_type(8)))  unsigned v8u;

// ---- cross-half-wave swap (lane ^ 16) ------------------------------------
__device__ __forceinline__ unsigned swap16u(unsigned v) {
#if __has_builtin(__builtin_amdgcn_permlanex16)
    // identity lane-select: lane i <- lane (i%16) of the opposite half
    return (unsigned)__builtin_amdgcn_permlanex16(
        (int)v, (int)v, 0x76543210, (int)0xfedcba98, false, false);
#else
    return (unsigned)__shfl_xor((int)v, 16, 32);
#endif
}
__device__ __forceinline__ float swap16f(float v) {
    return __builtin_bit_cast(float, swap16u(__builtin_bit_cast(unsigned, v)));
}

__device__ __forceinline__ float exp2fast(float x) {
#if __has_builtin(__builtin_amdgcn_exp2f)
    return __builtin_amdgcn_exp2f(x);
#else
    return exp2f(x);
#endif
}

// ---- async 16-byte global -> LDS copy (gfx1250 async path, ASYNCcnt) -----
__device__ __forceinline__ void async_cp16(unsigned lds_off, const float* g) {
    asm volatile("global_load_async_to_lds_b128 %0, %1, off"
                 :: "v"(lds_off), "v"((unsigned long long)(uintptr_t)g)
                 : "memory");
}
__device__ __forceinline__ void async_wait0() {
    asm volatile("s_wait_asynccnt 0" ::: "memory");
}

// ---------------------------------------------------------------------------
// Kernel 1: 1x1-conv projections. q,k -> [b][n][8] f16 ; v -> [b][c][n] f16.
// q is pre-scaled by log2(e) so attention softmax runs in exp2 domain.
// Weights staged into LDS via async global->LDS DMA.
// ---------------------------------------------------------------------------
__global__ void __launch_bounds__(256) proj_kernel(
    const float* __restrict__ x,
    const float* __restrict__ wq, const float* __restrict__ bq,
    const float* __restrict__ wk, const float* __restrict__ bk,
    const float* __restrict__ wv, const float* __restrict__ bv,
    _Float16* __restrict__ qh, _Float16* __restrict__ kh,
    _Float16* __restrict__ vt)
{
    // contiguous LDS block: wq | wk | wv | bq | bk | bv   (16B-aligned chunks)
    __shared__ float smem[NR*NC + NR*NC + NC*NC + NR + NR + NC];
    float* sWq = smem;
    float* sWk = smem + NR*NC;
    float* sWv = smem + 2*NR*NC;
    float* sB  = smem + 2*NR*NC + NC*NC;   // [bq(8) bk(8) bv(64)]

    const int tid = threadIdx.x;
    const unsigned l0 = (unsigned)(uintptr_t)&smem[0];

    if (tid < 128) {                                    // wq, wk: 512 f each
        async_cp16(l0 + (unsigned)(tid*16),               wq + tid*4);
        async_cp16(l0 + (unsigned)(NR*NC*4 + tid*16),     wk + tid*4);
    }
#pragma unroll
    for (int it = 0; it < 4; ++it) {                    // wv: 4096 f
        const int i = tid + it*256;
        async_cp16(l0 + (unsigned)(2*NR*NC*4 + i*16),     wv + i*4);
    }
    if (tid < 2)  async_cp16(l0 + (unsigned)((2*NR*NC + NC*NC)*4      + tid*16), bq + tid*4);
    if (tid < 2)  async_cp16(l0 + (unsigned)((2*NR*NC + NC*NC + 8)*4  + tid*16), bk + tid*4);
    if (tid < 16) async_cp16(l0 + (unsigned)((2*NR*NC + NC*NC + 16)*4 + tid*16), bv + tid*4);
    async_wait0();
    __syncthreads();

    const int p = blockIdx.x * 256 + tid;     // 0 .. NB*NPIX-1 (exact grid)
    const int b = p >> 12;
    const int n = p & (NPIX - 1);

    float xv[NC];
    const float* xb = x + ((size_t)b * NC) * NPIX + n;
#pragma unroll
    for (int c = 0; c < NC; ++c) xv[c] = xb[(size_t)c * NPIX];

    _Float16* qp = qh + (size_t)p * NR;
    _Float16* kp = kh + (size_t)p * NR;
#pragma unroll
    for (int r = 0; r < NR; ++r) {
        float aq = sB[r];
        float ak = sB[NR + r];
#pragma unroll
        for (int c = 0; c < NC; ++c) {
            aq += sWq[r * NC + c] * xv[c];
            ak += sWk[r * NC + c] * xv[c];
        }
        qp[r] = (_Float16)(aq * LOG2E);   // exp2-domain softmax downstream
        kp[r] = (_Float16)ak;
    }

#pragma unroll 4
    for (int o = 0; o < NC; ++o) {
        float a = sB[2 * NR + o];
#pragma unroll
        for (int c = 0; c < NC; ++c) a += sWv[o * NC + c] * xv[c];
        vt[((size_t)b * NC + o) * NPIX + n] = (_Float16)a;
    }
}

// ---------------------------------------------------------------------------
// Kernel 2: flash attention, one wave32 per (batch, 16-query tile).
// S^T = K^T Q (r padded 8->32) and O^T = V^T P^T via v_wmma_f32_16x16x32_f16.
// Per-lane online softmax in exp2 domain (each lane owns one query column).
// ---------------------------------------------------------------------------
__global__ void __launch_bounds__(32) attn_kernel(
    const float* __restrict__ x, const float* __restrict__ gamma,
    const _Float16* __restrict__ qh, const _Float16* __restrict__ kh,
    const _Float16* __restrict__ vt, float* __restrict__ out)
{
    const int tile  = blockIdx.x;          // 0 .. NB*256-1
    const int b     = tile >> 8;           // 256 query tiles per batch
    const int ibase = (tile & 255) << 4;
    const int lane  = threadIdx.x;         // wave32
    const bool lo   = lane < 16;
    const int  li   = lane & 15;

    // B operand for scores: Q block, K-dim = r padded to 32 with zeros.
    v16h bq16 = {};
    if (lo) {
        v8h qv = *(const v8h*)(qh + (size_t)(b * NPIX + ibase + li) * NR);
#pragma unroll
        for (int h = 0; h < 8; ++h) bq16[h] = qv[h];
    }

    v8f acc[4] = {};                       // O^T accumulators, 4 c-groups of 16
    const v8f zero8 = {};
    float m_i = -3.0e30f;                  // running max (log2 domain)
    float l_i = 0.0f;                      // running sum

    const _Float16* kb = kh + (size_t)b * NPIX * NR;
    const _Float16* vb = vt + (size_t)b * NC * NPIX;

    for (int j0 = 0; j0 < NPIX; j0 += 32) {
        // A operand: K rows j0..j0+15 / j0+16..j0+31, K-dim r padded to 32.
        v16h ak0 = {}, ak1 = {};
        if (lo) {
            v8h k0 = *(const v8h*)(kb + (size_t)(j0 + li) * NR);
            v8h k1 = *(const v8h*)(kb + (size_t)(j0 + 16 + li) * NR);
#pragma unroll
            for (int h = 0; h < 8; ++h) { ak0[h] = k0[h]; ak1[h] = k1[h]; }
        }

        // S^T tiles [16j x 16i]: VGPR g -> row j0+g (lanes<16) / j0+g+8.
        v8f s0 = __builtin_amdgcn_wmma_f32_16x16x32_f16(
            false, ak0, false, bq16, (short)0, zero8, false, false);
        v8f s1 = __builtin_amdgcn_wmma_f32_16x16x32_f16(
            false, ak1, false, bq16, (short)0, zero8, false, false);

        // Online softmax over this lane's query column (combine lane halves).
        float cm = -3.0e30f;
#pragma unroll
        for (int g = 0; g < 8; ++g) cm = fmaxf(cm, fmaxf(s0[g], s1[g]));
        cm = fmaxf(cm, swap16f(cm));
        const float mnew = fmaxf(m_i, cm);
        const float corr = exp2fast(m_i - mnew);
        m_i = mnew;

        float p0[8], p1[8];
        float rs = 0.0f;
#pragma unroll
        for (int g = 0; g < 8; ++g) {
            p0[g] = exp2fast(s0[g] - mnew);
            p1[g] = exp2fast(s1[g] - mnew);
            rs += p0[g] + p1[g];
        }
        rs += swap16f(rs);
        l_i = l_i * corr + rs;

#pragma unroll
        for (int cg = 0; cg < 4; ++cg)
#pragma unroll
            for (int g = 0; g < 8; ++g) acc[cg][g] *= corr;

        // Pack P^T as WMMA B operand (K=32 over j).  Pairs are packed to f16
        // first (v_cvt_pk_rz_f16_f32) so the half-wave exchange is 8 u32 ops.
        v8u bw;
#pragma unroll
        for (int w = 0; w < 4; ++w) {
            const unsigned u0 = __builtin_bit_cast(
                unsigned, __builtin_amdgcn_cvt_pkrtz(p0[2*w], p0[2*w+1]));
            const unsigned u1 = __builtin_bit_cast(
                unsigned, __builtin_amdgcn_cvt_pkrtz(p1[2*w], p1[2*w+1]));
            const unsigned x0 = swap16u(u0);
            const unsigned x1 = swap16u(u1);
            bw[w]     = lo ? u0 : x1;      // K = 2w,2w+1   / 16+2w,16+2w+1
            bw[4 + w] = lo ? x0 : u1;      // K = 8+2w,9+2w / 24+2w,25+2w
        }
        const v16h bp = __builtin_bit_cast(v16h, bw);

        // O^T += V^T * P^T  for each 16-channel group.
        const int jo = j0 + (lo ? 0 : 16); // A layout K-striping per half-wave
#pragma unroll
        for (int cg = 0; cg < 4; ++cg) {
            const _Float16* vp = vb + (size_t)(cg * 16 + li) * NPIX + jo;
            __builtin_prefetch(vp + 32, 0, 3);          // next j-chunk
            v8h a0 = *(const v8h*)(vp);                 // K 0..7   / 16..23
            v8h a1 = *(const v8h*)(vp + 8);             // K 8..15  / 24..31
            v16h av;
#pragma unroll
            for (int h = 0; h < 8; ++h) { av[h] = a0[h]; av[8 + h] = a1[h]; }
            acc[cg] = __builtin_amdgcn_wmma_f32_16x16x32_f16(
                false, av, false, bp, (short)0, acc[cg], false, false);
        }
    }

    // Epilogue: out = gamma * (O^T / l) + x, scattered per (c, i).
    const float inv  = 1.0f / l_i;
    const float gm   = gamma[0];
    const int   icol = ibase + li;
    const size_t xb0 = (size_t)b * NC * NPIX;
#pragma unroll
    for (int cg = 0; cg < 4; ++cg) {
#pragma unroll
        for (int g = 0; g < 8; ++g) {
            const int c = cg * 16 + g + (lo ? 0 : 8);
            const size_t idx = xb0 + (size_t)c * NPIX + icol;
            out[idx] = gm * (acc[cg][g] * inv) + x[idx];
        }
    }
}

// ---------------------------------------------------------------------------
extern "C" void kernel_launch(void* const* d_in, const int* in_sizes, int n_in,
                              void* d_out, int out_size, void* d_ws, size_t ws_size,
                              hipStream_t stream)
{
    (void)in_sizes; (void)n_in; (void)out_size; (void)ws_size;

    const float* x     = (const float*)d_in[0];
    const float* wq    = (const float*)d_in[1];
    const float* bq    = (const float*)d_in[2];
    const float* wk    = (const float*)d_in[3];
    const float* bk    = (const float*)d_in[4];
    const float* wv    = (const float*)d_in[5];
    const float* bv    = (const float*)d_in[6];
    const float* gamma = (const float*)d_in[7];
    float* out = (float*)d_out;

    _Float16* qh = (_Float16*)d_ws;                       // [B][N][8]   512 KB
    _Float16* kh = qh + (size_t)NB * NPIX * NR;           // [B][N][8]   512 KB
    _Float16* vt = kh + (size_t)NB * NPIX * NR;           // [B][C][N]     4 MB

    proj_kernel<<<(NB * NPIX) / 256, 256, 0, stream>>>(
        x, wq, bq, wk, bk, wv, bv, qh, kh, vt);

    attn_kernel<<<NB * (NPIX / 16), 32, 0, stream>>>(
        x, gamma, qh, kh, vt, out);
}
